// MH_U_MLP_79156247265938
// MI455X (gfx1250) — matrix operand order repaired and checked by reading, not compiled, not run on hardware
//
#include <hip/hip_runtime.h>
#include <math.h>

typedef __attribute__((ext_vector_type(16))) _Float16 v16h;
typedef __attribute__((ext_vector_type(8)))  float    v8f;

#define DEV static __device__ __forceinline__

// ---------------- scalar helpers ----------------
DEV unsigned short f32_to_h16(float f) {
  _Float16 h = (_Float16)f;
  return __builtin_bit_cast(unsigned short, h);
}
DEV float h16_to_f32(unsigned short u) {
  _Float16 h = __builtin_bit_cast(_Float16, u);
  return (float)h;
}
DEV float gelu_exact(float x) {
  return 0.5f * x * (1.0f + erff(x * 0.70710678118654752f));
}

union FragU { v16h v; uint4 q[2]; unsigned short s[16]; };

// A fragment: 16x32 f16, rows m0..m0+15, K cols k0..k0+31 (ISA 16-bit A layout)
DEV v16h load_a_frag(const unsigned short* lds, int stride, int m0, int k0) {
  int lane = threadIdx.x & 31;
  int row  = m0 + (lane & 15);
  int kb   = k0 + ((lane >> 4) << 3);          // + half*8
  const unsigned short* p = lds + row * stride + kb;
  FragU u;
  u.q[0] = *(const uint4*)(p);                 // K = kb .. kb+7
  u.q[1] = *(const uint4*)(p + 16);            // K = kb+16 .. kb+23
  return u.v;
}

// B fragment: 32x16 f16 from K-contiguous transposed tile Bt[n][k]
DEV v16h load_b_frag(const unsigned short* lds, int stride, int n0, int k0) {
  int lane = threadIdx.x & 31;
  int col  = n0 + (lane & 15);
  int kb   = k0 + ((lane >> 4) << 3);          // + half*8
  const unsigned short* p = lds + col * stride + kb;
  FragU u;
  u.q[0] = *(const uint4*)(p);
  u.q[1] = *(const uint4*)(p + 16);
  return u.v;
}

DEV v8f wmma_f16(v16h a, v16h b, v8f c) {
  return __builtin_amdgcn_wmma_f32_16x16x32_f16(false, a, false, b, (short)0, c,
                                                false, false);
}

// ---------------- Tensor Data Mover (2-D tile, f16 elements) ----------------
#if __has_builtin(__builtin_amdgcn_tensor_load_to_lds)
#define USE_TDM 1
typedef __attribute__((ext_vector_type(4))) unsigned int u32x4;
typedef __attribute__((ext_vector_type(8))) int i32x8;
typedef __attribute__((ext_vector_type(4))) int i32x4;

// Copy a [height x width] f16 tile (global row stride = row_stride elems) into
// LDS at byte offset lds_off, inserting pad_amt DWORDs after every pad_int
// DWORDs (encodes LDS row padding). Encodings per CDNA5 ISA D# group1.
// clang-23 toolchain: 6-arg form (g0 u32x4, g1 i32x8, g2 i32x4, g3 i32x4,
//                                 extra i32x8, cpol i32)
DEV void tdm_load_2d(unsigned lds_off, const void* gptr,
                     unsigned width, unsigned height, unsigned row_stride,
                     unsigned pad_int_enc, unsigned pad_amt_enc) {
  unsigned long long ga = (unsigned long long)gptr;
  u32x4 g0;
  g0.x = 1u;                                      // count=1, user descriptor
  g0.y = lds_off;                                 // LDS byte address
  g0.z = (unsigned)(ga & 0xFFFFFFFFu);            // global_addr[31:0]
  g0.w = (unsigned)((ga >> 32) & 0x1FFFFFFu) | (2u << 30);  // [56:32] | type=2
  i32x8 g1;
  g1[0] = (int)((1u << 16) |                      // data_size = 1 (2 bytes)
                (1u << 20) |                      // pad_enable
                (pad_int_enc << 22) | (pad_amt_enc << 25));
  g1[1] = (int)((row_stride & 0xFFFFu) << 16);    // tensor_dim0[15:0] (OOB = full row)
  g1[2] = (int)((row_stride >> 16) | ((height & 0xFFFFu) << 16)); // dim0 hi | dim1 lo
  g1[3] = (int)((height >> 16) | (width << 16));  // dim1 hi | tile_dim0
  g1[4] = (int)(height);                          // tile_dim1 (tile_dim2 = 0)
  g1[5] = (int)(row_stride);                      // tensor_dim0_stride[31:0]
  g1[6] = 0;                                      // stride hi | dim1_stride lo
  g1[7] = 0;
  i32x4 z4 = {0, 0, 0, 0};
  i32x8 z8 = {0, 0, 0, 0, 0, 0, 0, 0};
  __builtin_amdgcn_tensor_load_to_lds(g0, g1, z4, z4, z8, 0);
}
#else
#define USE_TDM 0
#endif

// ---------------- constants ----------------
#define BB 8
#define SS 2048
#define DD 1024
#define HN 16
#define HD 64
#define HH 256
#define EE 8
#define SD (SS*DD)            // 2,097,152

// ---------------- kernel 0: zero the logits accumulator ----------------
__global__ void k_zero(float* p, int n) {
  int i = blockIdx.x * blockDim.x + threadIdx.x;
  if (i < n) p[i] = 0.0f;
}

// ---------------- kernel 1: router partial dot products ----------------
__global__ __launch_bounds__(256) void k_router(const float* __restrict__ x,
                                                const float* __restrict__ rw,
                                                float* __restrict__ logits) {
  __shared__ float red[8][64];
  int t = threadIdx.x, wave = t >> 5, lane = t & 31;
  int base = blockIdx.x * 4096 + t * 16;

  float acc[8][8];
#pragma unroll
  for (int b = 0; b < 8; ++b)
#pragma unroll
    for (int e = 0; e < 8; ++e) acc[b][e] = 0.0f;

#pragma unroll
  for (int j = 0; j < 4; ++j) {
    int idx = base + j * 4;
    float4 xv[8];
#pragma unroll
    for (int b = 0; b < 8; ++b)
      xv[b] = *(const float4*)(x + (size_t)b * SD + idx);
#pragma unroll
    for (int e = 0; e < 8; ++e) {
      float4 wv = *(const float4*)(rw + (size_t)e * SD + idx);
#pragma unroll
      for (int b = 0; b < 8; ++b)
        acc[b][e] += xv[b].x * wv.x + xv[b].y * wv.y + xv[b].z * wv.z + xv[b].w * wv.w;
    }
  }
#pragma unroll
  for (int off = 16; off > 0; off >>= 1)
#pragma unroll
    for (int b = 0; b < 8; ++b)
#pragma unroll
      for (int e = 0; e < 8; ++e)
        acc[b][e] += __shfl_down(acc[b][e], off, 32);
  if (lane == 0)
#pragma unroll
    for (int b = 0; b < 8; ++b)
#pragma unroll
      for (int e = 0; e < 8; ++e) red[wave][b * 8 + e] = acc[b][e];
  __syncthreads();
  if (t < 64) {
    float s = 0.0f;
#pragma unroll
    for (int w = 0; w < 8; ++w) s += red[w][t];
    atomicAdd(&logits[t], s);
  }
}

// ---------------- kernel 2: softmax + top-2 ----------------
__global__ void k_topk(const float* __restrict__ logits, const float* __restrict__ rb,
                       float* __restrict__ gates, int* __restrict__ eidx) {
  int b = threadIdx.x;
  if (b >= BB) return;
  float l[8], mx = -1e30f;
#pragma unroll
  for (int e = 0; e < 8; ++e) { l[e] = logits[b * 8 + e] + rb[e]; mx = fmaxf(mx, l[e]); }
  float s = 0.0f;
#pragma unroll
  for (int e = 0; e < 8; ++e) { l[e] = expf(l[e] - mx); s += l[e]; }
  float inv = 1.0f / s;
  int i0 = 0; float p0 = -1.0f;
#pragma unroll
  for (int e = 0; e < 8; ++e) if (l[e] > p0) { p0 = l[e]; i0 = e; }
  int i1 = 0; float p1 = -1.0f;
#pragma unroll
  for (int e = 0; e < 8; ++e) if (e != i0 && l[e] > p1) { p1 = l[e]; i1 = e; }
  gates[b * 2 + 0] = p0 * inv; gates[b * 2 + 1] = p1 * inv;
  eidx[b * 2 + 0] = i0;        eidx[b * 2 + 1] = i1;
}

// ---------------- prep kernels: one-time f32 -> f16 conversions ----------------
// straight convert (x -> xh, w3 -> w3h); n4 = element_count/4
__global__ __launch_bounds__(256) void k_cvt(const float* __restrict__ src,
                                             unsigned short* __restrict__ dst, int n4) {
  int i = blockIdx.x * blockDim.x + threadIdx.x;
  if (i >= n4) return;
  float4 v = ((const float4*)src)[i];
  unsigned h0 = (unsigned)f32_to_h16(v.x) | ((unsigned)f32_to_h16(v.y) << 16);
  unsigned h1 = (unsigned)f32_to_h16(v.z) | ((unsigned)f32_to_h16(v.w) << 16);
  ((uint2*)dst)[i] = make_uint2(h0, h1);
}
// w1[e][d][f] -> w1h[e][f][d]   (B-fragment wants K(=d)-contiguous rows of f)
__global__ __launch_bounds__(256) void k_prep_w1(const float* __restrict__ w1,
                                                 unsigned short* __restrict__ w1h) {
  int i = blockIdx.x * 256 + threadIdx.x;        // i = e*16384 + d*256 + f
  int f = i & 255, d = (i >> 8) & 63, e = i >> 14;
  w1h[(e << 14) + f * 64 + d] = f32_to_h16(w1[i]);
}
// w2[e][f][d] -> w2h[e][d][f]
__global__ __launch_bounds__(256) void k_prep_w2(const float* __restrict__ w2,
                                                 unsigned short* __restrict__ w2h) {
  int i = blockIdx.x * 256 + threadIdx.x;        // i = e*16384 + f*64 + d
  int d = i & 63, f = (i >> 6) & 255, e = i >> 14;
  w2h[(e << 14) + d * 256 + f] = f32_to_h16(w2[i]);
}

// ---------------- kernel 3: fused expert MLP ----------------
// grid (S/128, H, B); 256 threads (8 waves). Each wave owns 16 rows.
// LDS: As[128][72], Hs[128][264], W1t[256][72], W2t[64][264]
#define AS_STR 72
#define HS_STR 264
#define LDS_AS   (128 * AS_STR)
#define LDS_HS   (128 * HS_STR)
#define LDS_W1   (256 * AS_STR)
#define LDS_W2   (64  * HS_STR)
#define MOE_LDS_U16 (LDS_AS + LDS_HS + LDS_W1 + LDS_W2)
#define OFF_AS_B  0
#define OFF_HS_B  (LDS_AS * 2)
#define OFF_W1_B  ((LDS_AS + LDS_HS) * 2)
#define OFF_W2_B  ((LDS_AS + LDS_HS + LDS_W1) * 2)

__global__ __launch_bounds__(256)
void k_moe(const unsigned short* __restrict__ xh,   // f16 [B,S,D]
           const unsigned short* __restrict__ w1h,  // f16 [E][HH][HD]
           const unsigned short* __restrict__ w2h,  // f16 [E][HD][HH]
           const float* __restrict__ b1, const float* __restrict__ b2,
           const float* __restrict__ gates, const int* __restrict__ eidx,
           unsigned short* __restrict__ y /* f16 [B,S,D] */) {
  extern __shared__ unsigned short smem[];
  unsigned short* As  = smem;
  unsigned short* Hs  = As + LDS_AS;
  unsigned short* W1t = Hs + LDS_HS;
  unsigned short* W2t = W1t + LDS_W1;

  int st = blockIdx.x, hh = blockIdx.y, b = blockIdx.z;
  int s0 = st * 128;
  int t = threadIdx.x, wave = t >> 5, lane = t & 31;
  int m0 = wave * 16;
  int hilo = (lane >> 4) * 8;
  int cl   = lane & 15;

  const unsigned short* xtile = xh + ((size_t)b * SS + s0) * DD + hh * HD;

#if USE_TDM
  if (t < 32)   // pad: insert 4 dwords (enc 3) every 32 dwords (enc 4) -> stride 72
    tdm_load_2d(OFF_AS_B, xtile, 64, 128, 1024, 4, 3);
#else
  {   // x tile (f16): 128 rows x 64, 2 threads/row, 4x uint4 each
    int r = t >> 1, kk = (t & 1) * 32;
    const uint4* src = (const uint4*)(xtile + (size_t)r * DD + kk);
    uint4* dst = (uint4*)(As + r * AS_STR + kk);
#pragma unroll
    for (int j = 0; j < 4; ++j) dst[j] = src[j];
  }
#endif

  float outacc[4][8];

#pragma unroll 1
  for (int k = 0; k < 2; ++k) {
    __syncthreads();   // prior iteration's W1t/W2t/Hs reads complete
    int e = eidx[b * 2 + k];
    float gate = gates[b * 2 + k];

#if USE_TDM
    if (t < 32) {
      tdm_load_2d(OFF_W1_B, w1h + ((size_t)e << 14), 64, 256, 64, 4, 3);
      tdm_load_2d(OFF_W2_B, w2h + ((size_t)e << 14), 256, 64, 256, 6, 3);
      __builtin_amdgcn_s_wait_tensorcnt(0);   // also covers the As load on k==0
    }
#else
    {   // W1t: 256 rows x 64
      int f = t >> 1, kk = (t & 1) * 32;
      const uint4* src = (const uint4*)(w1h + ((size_t)e << 14) + f * 64 + kk);
      uint4* dst = (uint4*)(W1t + f * AS_STR + kk);
#pragma unroll
      for (int j = 0; j < 4; ++j) dst[j] = src[j];
    }
    {   // W2t: 64 rows x 256
      int d = t >> 2, kk = (t & 3) * 64;
      const uint4* src = (const uint4*)(w2h + ((size_t)e << 14) + d * 256 + kk);
      uint4* dst = (uint4*)(W2t + d * HS_STR + kk);
#pragma unroll
      for (int j = 0; j < 8; ++j) dst[j] = src[j];
    }
#endif
    __syncthreads();

    if (k == 0) {   // residual init from the f16 x tile
#pragma unroll
      for (int nt = 0; nt < 4; ++nt)
#pragma unroll
        for (int i = 0; i < 8; ++i)
          outacc[nt][i] = h16_to_f32(As[(m0 + hilo + i) * AS_STR + nt * 16 + cl]);
    }

    // ---- GEMM1: [16x64] x [64x256] -> GELU -> Hs rows m0..m0+15 ----
    v16h a0 = load_a_frag(As, AS_STR, m0, 0);
    v16h a1 = load_a_frag(As, AS_STR, m0, 32);
#pragma unroll
    for (int nt = 0; nt < 16; ++nt) {
      int n0 = nt * 16;
      v8f acc = {0.f, 0.f, 0.f, 0.f, 0.f, 0.f, 0.f, 0.f};
      acc = wmma_f16(a0, load_b_frag(W1t, AS_STR, n0, 0),  acc);
      acc = wmma_f16(a1, load_b_frag(W1t, AS_STR, n0, 32), acc);
      float bias = b1[e * HH + n0 + cl];
      unsigned short* hp = Hs + (m0 + hilo) * HS_STR + n0 + cl;
#pragma unroll
      for (int i = 0; i < 8; ++i)
        hp[i * HS_STR] = f32_to_h16(gelu_exact(acc[i] + bias));
    }

    // ---- GEMM2: [16x256] x [256x64]; own rows only (no cross-wave dep) ----
    v8f acc2[4];
#pragma unroll
    for (int nt = 0; nt < 4; ++nt)
      acc2[nt] = (v8f){0.f, 0.f, 0.f, 0.f, 0.f, 0.f, 0.f, 0.f};
#pragma unroll
    for (int kk = 0; kk < 8; ++kk) {
      v16h af = load_a_frag(Hs, HS_STR, m0, kk * 32);
#pragma unroll
      for (int nt = 0; nt < 4; ++nt)
        acc2[nt] = wmma_f16(af, load_b_frag(W2t, HS_STR, nt * 16, kk * 32), acc2[nt]);
    }
#pragma unroll
    for (int nt = 0; nt < 4; ++nt) {
      float bias = b2[e * HD + nt * 16 + cl];
#pragma unroll
      for (int i = 0; i < 8; ++i)
        outacc[nt][i] += gate * (acc2[nt][i] + bias);
    }
  }

  // ---- write y = x + sum_k gate*o (f16) ----
  unsigned short* yp = y + ((size_t)b * SS + s0) * DD + hh * HD;
#pragma unroll
  for (int nt = 0; nt < 4; ++nt)
#pragma unroll
    for (int i = 0; i < 8; ++i)
      yp[(size_t)(m0 + hilo + i) * DD + nt * 16 + cl] = f32_to_h16(outacc[nt][i]);
}

// ---------------- kernel 4: out-projection [16384x1024] @ w3^T + b3 ----------------
__global__ __launch_bounds__(256)
void k_proj(const unsigned short* __restrict__ y, const unsigned short* __restrict__ w3h,
            const float* __restrict__ b3, float* __restrict__ out) {
  __shared__ unsigned short smemP[2 * 128 * 72];
  unsigned short* Ays = smemP;                 // y tile [row][k], LDS byte off 0
  unsigned short* Bws = smemP + 128 * 72;      // w3 tile [col][k], byte off 18432
  int c0 = blockIdx.x * 128;
  int r0 = blockIdx.y * 128;
  int t = threadIdx.x, wave = t >> 5, lane = t & 31;
  int m0 = wave * 16;
  int hilo = (lane >> 4) * 8, cl = lane & 15;

  v8f acc[8];
#pragma unroll
  for (int nt = 0; nt < 8; ++nt)
    acc[nt] = (v8f){0.f, 0.f, 0.f, 0.f, 0.f, 0.f, 0.f, 0.f};

#pragma unroll 1
  for (int k0 = 0; k0 < DD; k0 += 64) {
#if USE_TDM
    if (t < 32) {
      tdm_load_2d(0,     y   + (size_t)r0 * DD + k0, 64, 128, 1024, 4, 3);
      tdm_load_2d(18432, w3h + (size_t)c0 * DD + k0, 64, 128, 1024, 4, 3);
      __builtin_amdgcn_s_wait_tensorcnt(0);
    }
#else
    {   // y chunk
      int r = t >> 1, kk = (t & 1) * 32;
      const uint4* src = (const uint4*)(y + ((size_t)(r0 + r)) * DD + k0 + kk);
      uint4* dst = (uint4*)(Ays + r * 72 + kk);
#pragma unroll
      for (int j = 0; j < 4; ++j) dst[j] = src[j];
    }
    {   // w3 chunk (pre-converted f16)
      int c = t >> 1, kk = (t & 1) * 32;
      const uint4* src = (const uint4*)(w3h + ((size_t)(c0 + c)) * DD + k0 + kk);
      uint4* dst = (uint4*)(Bws + c * 72 + kk);
#pragma unroll
      for (int j = 0; j < 4; ++j) dst[j] = src[j];
    }
#endif
    if (k0 + 64 < DD) {   // prefetch next chunk (global_prefetch_b8)
      __builtin_prefetch(y   + ((size_t)(r0 + (t >> 1))) * DD + k0 + 64 + (t & 1) * 32, 0, 0);
      __builtin_prefetch(w3h + ((size_t)(c0 + (t >> 1))) * DD + k0 + 64 + (t & 1) * 32, 0, 0);
    }
    __syncthreads();
#pragma unroll
    for (int kk = 0; kk < 2; ++kk) {
      v16h af = load_a_frag(Ays, 72, m0, kk * 32);
#pragma unroll
      for (int nt = 0; nt < 8; ++nt)
        acc[nt] = wmma_f16(af, load_b_frag(Bws, 72, nt * 16, kk * 32), acc[nt]);
    }
    __syncthreads();
  }

#pragma unroll
  for (int nt = 0; nt < 8; ++nt) {
    int col = c0 + nt * 16 + cl;
    float bias = b3[col];
    float* op = out + (size_t)(r0 + m0 + hilo) * DD + col;
#pragma unroll
    for (int i = 0; i < 8; ++i) op[(size_t)i * DD] = acc[nt][i] + bias;
  }
}

// ---------------- host launcher ----------------
extern "C" void kernel_launch(void* const* d_in, const int* in_sizes, int n_in,
                              void* d_out, int out_size, void* d_ws, size_t ws_size,
                              hipStream_t stream) {
  (void)in_sizes; (void)n_in; (void)out_size; (void)ws_size;
  const float* x   = (const float*)d_in[0];
  const float* rw  = (const float*)d_in[1];
  const float* rb  = (const float*)d_in[2];
  const float* w1  = (const float*)d_in[3];
  const float* b1  = (const float*)d_in[4];
  const float* w2  = (const float*)d_in[5];
  const float* b2  = (const float*)d_in[6];
  const float* w3  = (const float*)d_in[7];
  const float* b3  = (const float*)d_in[8];
  float* out = (float*)d_out;

  char* ws = (char*)d_ws;
  float* logits = (float*)ws;                          // 64 f32
  float* gates  = (float*)(ws + 256);                  // 16 f32
  int*   eidx   = (int*)(ws + 512);                    // 16 i32
  size_t off = 4096;
  unsigned short* w1h = (unsigned short*)(ws + off); off += (size_t)EE * HH * HD * 2;
  unsigned short* w2h = (unsigned short*)(ws + off); off += (size_t)EE * HH * HD * 2;
  unsigned short* w3h = (unsigned short*)(ws + off); off += (size_t)DD * DD * 2;
  unsigned short* xh  = (unsigned short*)(ws + off); off += (size_t)BB * SS * DD * 2;
  unsigned short* y   = (unsigned short*)(ws + off);   // f16 [B,S,D]

  k_zero<<<1, 64, 0, stream>>>(logits, 64);
  k_router<<<512, 256, 0, stream>>>(x, rw, logits);
  k_topk<<<1, 32, 0, stream>>>(logits, rb, gates, eidx);

  k_cvt<<<(BB * SD / 4) / 256, 256, 0, stream>>>(x, xh, BB * SD / 4);
  k_cvt<<<(DD * DD / 4) / 256, 256, 0, stream>>>(w3, w3h, DD * DD / 4);
  k_prep_w1<<<(EE * HH * HD) / 256, 256, 0, stream>>>(w1, w1h);
  k_prep_w2<<<(EE * HH * HD) / 256, 256, 0, stream>>>(w2, w2h);

  size_t moe_lds = (size_t)MOE_LDS_U16 * sizeof(unsigned short);   // ~156 KB
  (void)hipFuncSetAttribute(reinterpret_cast<const void*>(k_moe),
                            hipFuncAttributeMaxDynamicSharedMemorySize, (int)moe_lds);
  k_moe<<<dim3(SS / 128, HN, BB), 256, moe_lds, stream>>>(xh, w1h, w2h, b1, b2,
                                                          gates, eidx, y);
  k_proj<<<dim3(DD / 128, (BB * SS) / 128), 256, 0, stream>>>(y, w3h, b3, out);
}
